// graphNetwork_try_18451179503911
// MI455X (gfx1250) — compile-verified, matrix operand order checked by reading, not run on hardware
//
#include <hip/hip_runtime.h>
#include <math.h>

// ---------------- problem constants (match reference) ----------------
#define N_NODES   4096
#define N_EDGES   65536
#define C_OPEN    128          // NOPEN == NHID == NCLOSE
#define NFEAT     384          // 3 * NOPEN
#define NLAYER    4
#define LIN_HID   256
#define NUM_OUT   1024
#define EPS_LN    1e-5f
#define H_STEP    0.1f
#define RED_BLOCKS 1024

typedef __attribute__((ext_vector_type(16))) __bf16 v16bf;
typedef __attribute__((ext_vector_type(8)))  __bf16 v8bf;
typedef __attribute__((ext_vector_type(8)))  float  v8f;
typedef __attribute__((ext_vector_type(4)))  unsigned int uint4v;
typedef __attribute__((ext_vector_type(8)))  int    int8v;
typedef __attribute__((ext_vector_type(4)))  int    int4v;

#if __has_builtin(__builtin_amdgcn_tensor_load_to_lds) && __has_builtin(__builtin_amdgcn_s_wait_tensorcnt)
#define USE_TDM 1
#else
#define USE_TDM 0
#endif

// Low 32 bits of a generic LDS pointer == wave-relative LDS byte offset.
__device__ __forceinline__ unsigned lds_offset_of(const void* p)
{
    return (unsigned)(unsigned long long)p;
}

// One TDM 2-D tile load: rows r=0..d1-1 of `base` (row stride `stride` bf16
// elems, d0 contiguous bf16 elems per row) -> LDS contiguous [d1][d0].
// Waits TENSORcnt==0.  Caller must __syncthreads().
__device__ __forceinline__ void tdm_load_2d(const __bf16* base, int d0, int d1,
                                            int stride, unsigned lds_byte_off)
{
#if USE_TDM
    const unsigned long long ga = (unsigned long long)(const void*)base;
    uint4v g0;
    g0[0] = 1u;                                    // count=1, user descriptor
    g0[1] = lds_byte_off;                          // lds_addr
    g0[2] = (unsigned)ga;                          // global_addr[31:0]
    g0[3] = (unsigned)((ga >> 32) & 0x1FFFFFFull) | (2u << 30);  // addr[56:32] | type=2
    int8v g1;
    g1[0] = (int)(1u << 16);                       // workgroup_mask=0, data_size=1 (2B)
    g1[1] = (int)(((unsigned)d0 & 0xFFFFu) << 16);               // tensor_dim0[15:0]
    g1[2] = (int)((((unsigned)d0 >> 16) & 0xFFFFu) |
                  (((unsigned)d1 & 0xFFFFu) << 16));             // dim0[31:16] | dim1[15:0]
    g1[3] = (int)(((unsigned)d0 & 0xFFFFu) << 16);               // dim1[31:16]=0 | tile_dim0=d0
    g1[4] = (int)((unsigned)d1 & 0xFFFFu);         // tile_dim1=d1, tile_dim2=0
    g1[5] = (int)(unsigned)stride;                 // tensor_dim0_stride[31:0]
    g1[6] = 0;                                     // stride0[47:32] | stride1[15:0]
    g1[7] = 0;                                     // stride1[47:16]
    int4v gz;  gz[0] = 0; gz[1] = 0; gz[2] = 0; gz[3] = 0;
    int8v gz8; gz8[0] = 0; gz8[1] = 0; gz8[2] = 0; gz8[3] = 0;
    gz8[4] = 0; gz8[5] = 0; gz8[6] = 0; gz8[7] = 0;
    __builtin_amdgcn_tensor_load_to_lds(g0, g1, gz, gz, gz8, 0);
    __builtin_amdgcn_s_wait_tensorcnt(0);
#else
    (void)base; (void)d0; (void)d1; (void)stride; (void)lds_byte_off;
#endif
}

// =====================================================================
// WMMA GEMM, channel-fastest layout:
//   Y[Nc, M] = X[Nc, K] * W[M, K]^T   (bf16 in, f32 accumulate)
// Block = 128 threads = 4 waves; block tile 32(rows n) x 32(cols m).
// A-fragment: activation rows, K-contiguous -> two global b128 per lane.
// B-fragment: weight rows, K-contiguous -> staged once per block into LDS
// with ONE tensor_load_to_lds DMA (rows m0..m0+31, all K), then two
// ds_load_b128 per lane per k-step.  No per-k-step barriers.
// Optional bias + ELU, optional fp32 and/or bf16 outputs.
// =====================================================================
__global__ __launch_bounds__(128)
void gemm_wmma_bf16(const __bf16* __restrict__ W, const __bf16* __restrict__ X,
                    float* __restrict__ Yf, __bf16* __restrict__ Yb,
                    const float* __restrict__ bias,
                    int M, int K, int Nrows, int act)
{
    __shared__ __bf16 ws_[32 * NFEAT];             // up to 24 KB weight strip
    const int tid  = threadIdx.x;
    const int lane = tid & 31;
    const int wid  = tid >> 5;
    const int wm   = wid >> 1;                     // wave row tile (0..1)
    const int wn   = wid & 1;                      // wave col tile (0..1)
    const int m0   = blockIdx.x * 32;              // output-channel tile
    const int n0   = blockIdx.y * 32;              // row tile
    const int r    = lane & 15;
    const int kb   = (lane >> 4) * 8;              // ISA 16-bit fragment K base

#if USE_TDM
    if (wid == 0)
        tdm_load_2d(W + (size_t)m0 * K, K, 32, K, lds_offset_of(ws_));
#else
    for (int e = tid * 8; e < 32 * K; e += 128 * 8) {
        const int row = e / K, k = e % K;
        *(v8bf*)(ws_ + e) = *(const v8bf*)(W + (size_t)(m0 + row) * K + k);
    }
#endif
    __syncthreads();

    const __bf16* ap = X   + (size_t)(n0 + wm * 16 + r) * K;   // activation row
    const __bf16* bp = ws_ + (size_t)(wn * 16 + r) * K;        // weight row (LDS)

    v8f acc = {};
    const int ksteps = K >> 5;
    for (int kk = 0; kk < ksteps; ++kk) {
        const int ko = kk * 32 + kb;
        const v8bf a0 = *(const v8bf*)(ap + ko);
        const v8bf a1 = *(const v8bf*)(ap + ko + 16);
        const v8bf b0 = *(const v8bf*)(bp + ko);
        const v8bf b1 = *(const v8bf*)(bp + ko + 16);
        v16bf a, b;
#pragma unroll
        for (int h = 0; h < 8; ++h) { a[h] = a0[h]; a[8 + h] = a1[h];
                                      b[h] = b0[h]; b[8 + h] = b1[h]; }
        acc = __builtin_amdgcn_wmma_f32_16x16x32_bf16(false, a, false, b,
                                                      (short)0, acc, false, false);
    }
    // epilogue: D rows = activation rows (n), cols = output channels (m)
    const int roff = (lane >> 4) * 8;
    const int m    = m0 + wn * 16 + (lane & 15);
#pragma unroll
    for (int i = 0; i < 8; ++i) {
        const int n = n0 + wm * 16 + roff + i;
        float y = acc[i];
        if (bias) y += bias[m];
        if (act == 1) y = (y > 0.f) ? y : (expf(y) - 1.f);   // ELU
        const size_t o = (size_t)n * M + m;
        if (Yf) Yf[o] = y;
        if (Yb) Yb[o] = (__bf16)y;
    }
}

// =====================================================================
// Fused pairwise-distance statistics (channel-fastest node features):
//   G = fn * fn^T, fn = [N,128] bf16; D_ij = relu(s_i + s_j - 2 G_ij).
// Both WMMA operands are rows of fn -> pure contiguous global b128 loads,
// no LDS staging, no transpose kernel.  D never materialized: per-block
// sum/sumsq -> partials[2*bid].
// =====================================================================
__global__ __launch_bounds__(128)
void dmat_stats_wmma(const __bf16* __restrict__ fn, const float* __restrict__ s,
                     float* __restrict__ partials)
{
    __shared__ float r0[128], r1[128];
    const int tid  = threadIdx.x;
    const int lane = tid & 31;
    const int wid  = tid >> 5;
    const int wm   = wid >> 1;
    const int wn   = wid & 1;
    const int j0   = blockIdx.x * 32;
    const int i0   = blockIdx.y * 32;
    const int r    = lane & 15;
    const int kb   = (lane >> 4) * 8;

    const __bf16* ap = fn + (size_t)(i0 + wm * 16 + r) * C_OPEN;
    const __bf16* bp = fn + (size_t)(j0 + wn * 16 + r) * C_OPEN;

    v8f acc = {};
#pragma unroll
    for (int kk = 0; kk < 4; ++kk) {               // K = 128
        const int ko = kk * 32 + kb;
        const v8bf a0 = *(const v8bf*)(ap + ko);
        const v8bf a1 = *(const v8bf*)(ap + ko + 16);
        const v8bf b0 = *(const v8bf*)(bp + ko);
        const v8bf b1 = *(const v8bf*)(bp + ko + 16);
        v16bf a, b;
#pragma unroll
        for (int h = 0; h < 8; ++h) { a[h] = a0[h]; a[8 + h] = a1[h];
                                      b[h] = b0[h]; b[8 + h] = b1[h]; }
        acc = __builtin_amdgcn_wmma_f32_16x16x32_bf16(false, a, false, b,
                                                      (short)0, acc, false, false);
    }
    // fused epilogue: D = relu(s_i + s_j - 2g); accumulate stats only
    const int roff = (lane >> 4) * 8;
    const int gj   = j0 + wn * 16 + (lane & 15);
    float lsum = 0.f, lsq = 0.f;
    const float sj = s[gj];
#pragma unroll
    for (int i = 0; i < 8; ++i) {
        const int gi = i0 + wm * 16 + roff + i;
        float d = s[gi] + sj - 2.f * acc[i];
        d = fmaxf(d, 0.f);
        lsum += d; lsq += d * d;
    }
    r0[tid] = lsum; r1[tid] = lsq;
    __syncthreads();
    for (int st = 64; st > 0; st >>= 1) {
        if (tid < st) { r0[tid] += r0[tid + st]; r1[tid] += r1[tid + st]; }
        __syncthreads();
    }
    if (tid == 0) {
        const int bid = blockIdx.y * gridDim.x + blockIdx.x;
        partials[2 * bid]     = r0[0];
        partials[2 * bid + 1] = r1[0];
    }
}

// ---------------- two-stage deterministic reductions ----------------
__global__ __launch_bounds__(256)
void reduce_sum2(const float* __restrict__ x, int n, float* __restrict__ partials)
{
    __shared__ float r0[256], r1[256];
    float a = 0.f, b = 0.f;
    for (int i = blockIdx.x * blockDim.x + threadIdx.x; i < n; i += gridDim.x * blockDim.x) {
        const float v = x[i]; a += v; b += v * v;
    }
    r0[threadIdx.x] = a; r1[threadIdx.x] = b;
    __syncthreads();
    for (int st = 128; st > 0; st >>= 1) {
        if (threadIdx.x < st) { r0[threadIdx.x] += r0[threadIdx.x + st];
                                r1[threadIdx.x] += r1[threadIdx.x + st]; }
        __syncthreads();
    }
    if (threadIdx.x == 0) {
        partials[2 * blockIdx.x]     = r0[0];
        partials[2 * blockIdx.x + 1] = r1[0];
    }
}

__global__ __launch_bounds__(256)
void reduce_finalize(const float* __restrict__ partials, int nparts,
                     float count, float ddof, float* __restrict__ stats)
{
    __shared__ float r0[256], r1[256];
    float a = 0.f, b = 0.f;
    for (int i = threadIdx.x; i < nparts; i += blockDim.x) {
        a += partials[2 * i]; b += partials[2 * i + 1];
    }
    r0[threadIdx.x] = a; r1[threadIdx.x] = b;
    __syncthreads();
    for (int st = 128; st > 0; st >>= 1) {
        if (threadIdx.x < st) { r0[threadIdx.x] += r0[threadIdx.x + st];
                                r1[threadIdx.x] += r1[threadIdx.x + st]; }
        __syncthreads();
    }
    if (threadIdx.x == 0) {
        const float mean = r0[0] / count;
        float var = (r1[0] - count * mean * mean) / (count - ddof);
        stats[0] = mean;
        stats[1] = fmaxf(var, 0.f);
    }
}

// ---------------- elementwise kernels ----------------
__global__ void cvt_bf16_kernel(const float* __restrict__ x, __bf16* __restrict__ y, int n)
{
    const int i = blockIdx.x * blockDim.x + threadIdx.x;
    if (i < n) y[i] = (__bf16)x[i];
}

// input [64, Nc] channel-major fp32 -> [Nc, 64] channel-fastest bf16
__global__ void cvt_t64_bf16_kernel(const float* __restrict__ x, __bf16* __restrict__ y,
                                    int Nc)
{
    const int t = blockIdx.x * blockDim.x + threadIdx.x;
    if (t >= Nc * 64) return;
    const int c = t & 63;
    const int n = t >> 6;
    y[t] = (__bf16)x[(size_t)c * Nc + n];
}

// y = bf16( tanh( (x - mean) * rsqrt(var + eps) ) )   [inner ln of double_layer]
__global__ void ln_tanh_bf16_kernel(const float* __restrict__ x,
                                    const float* __restrict__ stats,
                                    __bf16* __restrict__ y, int n)
{
    const int i = blockIdx.x * blockDim.x + threadIdx.x;
    if (i >= n) return;
    const float rs = rsqrtf(stats[1] + EPS_LN);
    y[i] = (__bf16)tanhf((x[i] - stats[0]) * rs);
}

// xe += H * ln_all(g); refresh bf16 shadow
__global__ void xe_update_kernel(float* __restrict__ xe, const float* __restrict__ g,
                                 const float* __restrict__ stats,
                                 __bf16* __restrict__ xeb, int n)
{
    const int i = blockIdx.x * blockDim.x + threadIdx.x;
    if (i >= n) return;
    const float rs = rsqrtf(stats[1] + EPS_LN);
    const float v  = xe[i] + H_STEP * ((g[i] - stats[0]) * rs);
    xe[i]  = v;
    xeb[i] = (__bf16)v;
}

// xn += H * g; refresh bf16 shadow
__global__ void xn_update_kernel(float* __restrict__ xn, const float* __restrict__ g,
                                 __bf16* __restrict__ xnb, int n)
{
    const int i = blockIdx.x * blockDim.x + threadIdx.x;
    if (i >= n) return;
    const float v = xn[i] + H_STEP * g[i];
    xn[i]  = v;
    xnb[i] = (__bf16)v;
}

__global__ void zero_f32_kernel(float* __restrict__ p, int n)
{
    const int i = blockIdx.x * blockDim.x + threadIdx.x;
    if (i < n) p[i] = 0.f;
}

// s[n] = |f_n|^2  (contiguous row sum, xn is [N,128])
__global__ void node_sq_kernel(const float* __restrict__ xn, float* __restrict__ s)
{
    const int n = blockIdx.x * blockDim.x + threadIdx.x;
    if (n >= N_NODES) return;
    const float* row = xn + (size_t)n * C_OPEN;
    float a = 0.f;
    for (int c = 0; c < C_OPEN; ++c) { const float v = row[c]; a += v * v; }
    s[n] = a;
}

// w[e] = exp(-2 * relu(s_i + s_j - 2 f_i.f_j) / std(D))   (row-contiguous dots)
__global__ void edge_w_kernel(const float* __restrict__ xn, const int* __restrict__ I,
                              const int* __restrict__ J, const float* __restrict__ s,
                              const float* __restrict__ statsD, float* __restrict__ w)
{
    const int e = blockIdx.x * blockDim.x + threadIdx.x;
    if (e >= N_EDGES) return;
    const int i = I[e], j = J[e];
    const float* ri = xn + (size_t)i * C_OPEN;
    const float* rj = xn + (size_t)j * C_OPEN;
    float dot = 0.f;
    for (int c = 0; c < C_OPEN; ++c) dot += ri[c] * rj[c];
    const float d    = fmaxf(s[i] + s[j] - 2.f * dot, 0.f);
    const float stdv = sqrtf(statsD[1]);
    w[e] = expf(-2.f * d / stdv);
}

// dxe[e] = [0.5*w*(xn_I + xn_J) | xe | w*(xn_I - xn_J)]  -> bf16 [E, 384]
__global__ void build_dxe_kernel(const float* __restrict__ xn, const float* __restrict__ xe,
                                 const float* __restrict__ w, const int* __restrict__ I,
                                 const int* __restrict__ J, __bf16* __restrict__ dxe)
{
    const int t = blockIdx.x * blockDim.x + threadIdx.x;
    if (t >= N_EDGES * C_OPEN) return;
    const int c = t & (C_OPEN - 1);
    const int e = t >> 7;
    const float a  = xn[(size_t)I[e] * C_OPEN + c];
    const float b  = xn[(size_t)J[e] * C_OPEN + c];
    const float wv = w[e];
    __bf16* row = dxe + (size_t)e * NFEAT;
    row[c]               = (__bf16)(0.5f * wv * (a + b));          // intX
    row[C_OPEN + c]      = (__bf16)xe[(size_t)e * C_OPEN + c];
    row[2 * C_OPEN + c]  = (__bf16)(wv * (a - b));                 // gradX
}

// divE / aveE scatter-adds; lanes cover one node row -> contiguous atomics
__global__ void edge_scatter_kernel(const float* __restrict__ xe, const float* __restrict__ w,
                                    const int* __restrict__ I, const int* __restrict__ J,
                                    float* __restrict__ divE, float* __restrict__ aveE)
{
    const int t = blockIdx.x * blockDim.x + threadIdx.x;
    if (t >= N_EDGES * C_OPEN) return;
    const int c = t & (C_OPEN - 1);
    const int e = t >> 7;
    const float wg = w[e] * xe[(size_t)e * C_OPEN + c];
    const int vi = I[e], vj = J[e];
    atomicAdd(&divE[(size_t)vi * C_OPEN + c],  wg);
    atomicAdd(&divE[(size_t)vj * C_OPEN + c], -wg);
    atomicAdd(&aveE[(size_t)vi * C_OPEN + c], 0.5f * wg);
    atomicAdd(&aveE[(size_t)vj * C_OPEN + c], 0.5f * wg);
}

// dxn[n] = [aveE | divE | xn] -> bf16 [N, 384]
__global__ void build_dxn_kernel(const float* __restrict__ aveE, const float* __restrict__ divE,
                                 const float* __restrict__ xn, __bf16* __restrict__ dxn)
{
    const int t = blockIdx.x * blockDim.x + threadIdx.x;
    if (t >= N_NODES * C_OPEN) return;
    const int c = t & (C_OPEN - 1);
    const int n = t >> 7;
    __bf16* row = dxn + (size_t)n * NFEAT;
    row[c]              = (__bf16)aveE[t];
    row[C_OPEN + c]     = (__bf16)divE[t];
    row[2 * C_OPEN + c] = (__bf16)xn[t];
}

// row-wise log_softmax on logits [N, 1024]; out [N, 1024]
__global__ __launch_bounds__(256)
void logsoftmax_kernel(const float* __restrict__ logits, float* __restrict__ out)
{
    __shared__ float red[256];
    const int n = blockIdx.x;
    const int t = threadIdx.x;
    const float* row = logits + (size_t)n * NUM_OUT;
    float mx = -3.4e38f;
    for (int o = t; o < NUM_OUT; o += 256) mx = fmaxf(mx, row[o]);
    red[t] = mx; __syncthreads();
    for (int st = 128; st > 0; st >>= 1) {
        if (t < st) red[t] = fmaxf(red[t], red[t + st]);
        __syncthreads();
    }
    mx = red[0]; __syncthreads();
    float sm = 0.f;
    for (int o = t; o < NUM_OUT; o += 256) sm += expf(row[o] - mx);
    red[t] = sm; __syncthreads();
    for (int st = 128; st > 0; st >>= 1) {
        if (t < st) red[t] += red[t + st];
        __syncthreads();
    }
    const float lse = mx + logf(red[0]);
    for (int o = t; o < NUM_OUT; o += 256)
        out[(size_t)n * NUM_OUT + o] = row[o] - lse;
}

// =====================================================================
// Host-side orchestration (all tensors channel-fastest: [rows, channels])
// =====================================================================
extern "C" void kernel_launch(void* const* d_in, const int* in_sizes, int n_in,
                              void* d_out, int out_size, void* d_ws, size_t ws_size,
                              hipStream_t stream)
{
    (void)in_sizes; (void)n_in; (void)out_size; (void)ws_size;

    const float* xn_in   = (const float*)d_in[0];
    const float* xe_in   = (const float*)d_in[1];
    const int*   iInd    = (const int*)  d_in[2];
    const int*   jInd    = (const int*)  d_in[3];
    const float* K1Nopen = (const float*)d_in[4];
    const float* K2Nopen = (const float*)d_in[5];
    const float* K1Eopen = (const float*)d_in[6];
    const float* K2Eopen = (const float*)d_in[7];
    const float* KNclose = (const float*)d_in[8];
    const float* KE1     = (const float*)d_in[9];
    const float* KE2     = (const float*)d_in[10];
    const float* KN1     = (const float*)d_in[11];
    const float* KN2     = (const float*)d_in[12];
    const float* W1      = (const float*)d_in[13];
    const float* b1      = (const float*)d_in[14];
    const float* W2      = (const float*)d_in[15];
    const float* b2      = (const float*)d_in[16];

    const int CN = N_NODES * C_OPEN;     // 524288
    const int CE = N_EDGES * C_OPEN;     // 8388608

    // ---- workspace carve-up ----
    char* base = (char*)d_ws;
    size_t off = 0;
    auto alloc = [&](size_t bytes) -> char* {
        char* r = base + off;
        off = (off + bytes + 255) & ~(size_t)255;
        return r;
    };
    float*  xnF   = (float*) alloc((size_t)CN * 4);     // [N,128]
    __bf16* xnB   = (__bf16*)alloc((size_t)CN * 2);
    float*  xeF   = (float*) alloc((size_t)CE * 4);     // [E,128]
    __bf16* xeB   = (__bf16*)alloc((size_t)CE * 2);
    __bf16* dxeB  = (__bf16*)alloc((size_t)N_EDGES * NFEAT * 2);   // [E,384]
    float*  t32E  = (float*) alloc((size_t)CE * 4);     // reused for 2nd gemm output
    __bf16* tbfE  = (__bf16*)alloc((size_t)CE * 2);
    __bf16* dxnB  = (__bf16*)alloc((size_t)N_NODES * NFEAT * 2);   // [N,384]
    float*  t32N  = (float*) alloc((size_t)CN * 4);
    __bf16* tbfN  = (__bf16*)alloc((size_t)CN * 2);
    float*  sN    = (float*) alloc((size_t)N_NODES * 4);
    float*  wE    = (float*) alloc((size_t)N_EDGES * 4);
    float*  divE  = (float*) alloc((size_t)CN * 4);
    float*  aveE  = (float*) alloc((size_t)CN * 4);
    float*  part  = (float*) alloc((size_t)16384 * 2 * 4);
    float*  stats = (float*) alloc(64);
    __bf16* xinB  = (__bf16*)alloc((size_t)N_EDGES * 64 * 2);      // input cvt scratch
    __bf16* xncB  = (__bf16*)alloc((size_t)CN * 2);
    __bf16* hB    = (__bf16*)alloc((size_t)N_NODES * LIN_HID * 2);
    float*  logit = (float*) alloc((size_t)N_NODES * NUM_OUT * 4);
    // bf16 weight shadows (converted once per call)
    __bf16* K1NB = (__bf16*)alloc((size_t)C_OPEN * 64 * 2);
    __bf16* K2NB = (__bf16*)alloc((size_t)C_OPEN * C_OPEN * 2);
    __bf16* K1EB = (__bf16*)alloc((size_t)C_OPEN * 64 * 2);
    __bf16* K2EB = (__bf16*)alloc((size_t)C_OPEN * C_OPEN * 2);
    __bf16* KNcB = (__bf16*)alloc((size_t)C_OPEN * C_OPEN * 2);
    __bf16* KE1B = (__bf16*)alloc((size_t)NLAYER * C_OPEN * NFEAT * 2);
    __bf16* KE2B = (__bf16*)alloc((size_t)NLAYER * C_OPEN * C_OPEN * 2);
    __bf16* KN1B = (__bf16*)alloc((size_t)NLAYER * C_OPEN * NFEAT * 2);
    __bf16* KN2B = (__bf16*)alloc((size_t)NLAYER * C_OPEN * C_OPEN * 2);
    __bf16* W1B  = (__bf16*)alloc((size_t)LIN_HID * C_OPEN * 2);
    __bf16* W2B  = (__bf16*)alloc((size_t)NUM_OUT * LIN_HID * 2);
    float*  outF = (float*)d_out;

    auto ew = [&](int n) { return dim3((n + 255) / 256); };
    auto cvt = [&](const float* src, __bf16* dst, int n) {
        cvt_bf16_kernel<<<ew(n), 256, 0, stream>>>(src, dst, n);
    };
    // Y[Nrows, M] = X[Nrows, K] * W[M, K]^T
    auto gemm = [&](const __bf16* W, const __bf16* X, float* Yf, __bf16* Yb,
                    const float* bias, int M, int K, int Nrows, int act) {
        dim3 g(M / 32, Nrows / 32);
        gemm_wmma_bf16<<<g, dim3(128), 0, stream>>>(W, X, Yf, Yb, bias, M, K, Nrows, act);
    };
    auto stats_of = [&](const float* x, int n, float ddof) {
        reduce_sum2<<<dim3(RED_BLOCKS), dim3(256), 0, stream>>>(x, n, part);
        reduce_finalize<<<dim3(1), dim3(256), 0, stream>>>(part, RED_BLOCKS, (float)n, ddof, stats);
    };

    // ---- one-time (per call) weight downconversions ----
    cvt(K1Nopen, K1NB, C_OPEN * 64);
    cvt(K2Nopen, K2NB, C_OPEN * C_OPEN);
    cvt(K1Eopen, K1EB, C_OPEN * 64);
    cvt(K2Eopen, K2EB, C_OPEN * C_OPEN);
    cvt(KNclose, KNcB, C_OPEN * C_OPEN);
    cvt(KE1, KE1B, NLAYER * C_OPEN * NFEAT);
    cvt(KE2, KE2B, NLAYER * C_OPEN * C_OPEN);
    cvt(KN1, KN1B, NLAYER * C_OPEN * NFEAT);
    cvt(KN2, KN2B, NLAYER * C_OPEN * C_OPEN);
    cvt(W1, W1B, LIN_HID * C_OPEN);
    cvt(W2, W2B, NUM_OUT * LIN_HID);

    // ================= open double-layers =================
    // nodes: [N,64] -> [N,128]
    cvt_t64_bf16_kernel<<<ew(64 * N_NODES), 256, 0, stream>>>(xn_in, xinB, N_NODES);
    gemm(K1NB, xinB, t32N, nullptr, nullptr, C_OPEN, 64, N_NODES, 0);
    stats_of(t32N, CN, 0.f);
    ln_tanh_bf16_kernel<<<ew(CN), 256, 0, stream>>>(t32N, stats, tbfN, CN);
    gemm(K2NB, tbfN, xnF, xnB, nullptr, C_OPEN, C_OPEN, N_NODES, 0);

    // edges: [E,64] -> [E,128]
    cvt_t64_bf16_kernel<<<ew(64 * N_EDGES), 256, 0, stream>>>(xe_in, xinB, N_EDGES);
    gemm(K1EB, xinB, t32E, nullptr, nullptr, C_OPEN, 64, N_EDGES, 0);
    stats_of(t32E, CE, 0.f);
    ln_tanh_bf16_kernel<<<ew(CE), 256, 0, stream>>>(t32E, stats, tbfE, CE);
    gemm(K2EB, tbfE, xeF, xeB, nullptr, C_OPEN, C_OPEN, N_EDGES, 0);

    // ================= message-passing layers =================
    for (int l = 0; l < NLAYER; ++l) {
        const __bf16* ke1 = KE1B + (size_t)l * C_OPEN * NFEAT;
        const __bf16* ke2 = KE2B + (size_t)l * C_OPEN * C_OPEN;
        const __bf16* kn1 = KN1B + (size_t)l * C_OPEN * NFEAT;
        const __bf16* kn2 = KN2B + (size_t)l * C_OPEN * C_OPEN;

        // --- fused pairwise-distance stats (D never materialized) ---
        node_sq_kernel<<<ew(N_NODES), 256, 0, stream>>>(xnF, sN);
        dmat_stats_wmma<<<dim3(N_NODES / 32, N_NODES / 32), dim3(128), 0, stream>>>(xnB, sN, part);
        reduce_finalize<<<dim3(1), dim3(256), 0, stream>>>(
            part, 16384, (float)N_NODES * (float)N_NODES, 1.f, stats);   // ddof=1 (torch .std)
        edge_w_kernel<<<ew(N_EDGES), 256, 0, stream>>>(xnF, iInd, jInd, sN, stats, wE);

        // --- edge update: dxe = ln(double_layer(concat)); xe += H*dxe ---
        build_dxe_kernel<<<ew(CE), 256, 0, stream>>>(xnF, xeF, wE, iInd, jInd, dxeB);
        gemm(ke1, dxeB, t32E, nullptr, nullptr, C_OPEN, NFEAT, N_EDGES, 0);
        stats_of(t32E, CE, 0.f);
        ln_tanh_bf16_kernel<<<ew(CE), 256, 0, stream>>>(t32E, stats, tbfE, CE);
        gemm(ke2, tbfE, t32E, nullptr, nullptr, C_OPEN, C_OPEN, N_EDGES, 0);
        stats_of(t32E, CE, 0.f);
        xe_update_kernel<<<ew(CE), 256, 0, stream>>>(xeF, t32E, stats, xeB, CE);

        // --- node update: scatter div/ave, then double_layer; xn += H*dxn ---
        zero_f32_kernel<<<ew(CN), 256, 0, stream>>>(divE, CN);
        zero_f32_kernel<<<ew(CN), 256, 0, stream>>>(aveE, CN);
        edge_scatter_kernel<<<ew(CE), 256, 0, stream>>>(xeF, wE, iInd, jInd, divE, aveE);
        build_dxn_kernel<<<ew(CN), 256, 0, stream>>>(aveE, divE, xnF, dxnB);
        gemm(kn1, dxnB, t32N, nullptr, nullptr, C_OPEN, NFEAT, N_NODES, 0);
        stats_of(t32N, CN, 0.f);
        ln_tanh_bf16_kernel<<<ew(CN), 256, 0, stream>>>(t32N, stats, tbfN, CN);
        gemm(kn2, tbfN, t32N, nullptr, nullptr, C_OPEN, C_OPEN, N_NODES, 0);
        xn_update_kernel<<<ew(CN), 256, 0, stream>>>(xnF, t32N, xnB, CN);
    }

    // ================= close + MLP head =================
    gemm(KNcB, xnB, nullptr, xncB, nullptr, C_OPEN, C_OPEN, N_NODES, 0);          // [N,128]
    gemm(W1B, xncB, nullptr, hB, b1, LIN_HID, C_OPEN, N_NODES, 1);                // ELU, [N,256]
    gemm(W2B, hB, logit, nullptr, b2, NUM_OUT, LIN_HID, N_NODES, 0);              // [N,1024]
    logsoftmax_kernel<<<dim3(N_NODES), dim3(256), 0, stream>>>(logit, outF);      // row-wise
}
